// SeqEmbed_34660386079025
// MI455X (gfx1250) — matrix operand order repaired and checked by reading, not compiled, not run on hardware
//
#include <hip/hip_runtime.h>
#include <hip/hip_bf16.h>

typedef __attribute__((ext_vector_type(16))) _Float16 v16h;
typedef __attribute__((ext_vector_type(8)))  _Float16 v8h;
typedef __attribute__((ext_vector_type(4)))  _Float16 v4h;
typedef __attribute__((ext_vector_type(8)))  float    v8f;
typedef __attribute__((ext_vector_type(4)))  float    v4f;

#define L 384
#define D 32
#define SEQ_E 256
#define P 128
#define NBS 10
#define NBP 65
#define SEQH_STRIDE 40   // padded halves per row (80B, 16B-aligned, conflict-free)
#define T_STRIDE 40

// ---------------- prologue: PWT[rel][p] = pos_pair_embed_w[p][rel] ----------------
__global__ void pwt_kernel(const float* __restrict__ pos_pair_w, float* __restrict__ PWT) {
    for (int idx = threadIdx.x; idx < NBP * P; idx += blockDim.x) {
        int rel = idx / P, p = idx % P;
        PWT[idx] = pos_pair_w[p * NBP + rel];
    }
}

// ---------------- prologue: SPI[j][p], SPJ'[j][p] (biases folded) ----------------
__global__ void sp_kernel(const float* __restrict__ seq,
                          const float* __restrict__ spi_w, const float* __restrict__ spi_b,
                          const float* __restrict__ spj_w, const float* __restrict__ spj_b,
                          const float* __restrict__ W_pair_b, const float* __restrict__ pos_pair_b,
                          float* __restrict__ SPI, float* __restrict__ SPJc) {
    __shared__ float srow[D];
    int j = blockIdx.x, p = threadIdx.x;
    if (p < D) srow[p] = seq[j * D + p];
    __syncthreads();
    float a1 = spi_b[p];
    float a2 = spj_b[p] + W_pair_b[p] + pos_pair_b[p];
    const float* wi = spi_w + p * D;
    const float* wj = spj_w + p * D;
#pragma unroll
    for (int m = 0; m < D; ++m) { a1 += srow[m] * wi[m]; a2 += srow[m] * wj[m]; }
    SPI[j * P + p]  = a1;
    SPJc[j * P + p] = a2;
}

// ---------------- prologue: seq_embed output (tiny, scalar FMA) ----------------
__global__ void seq_embed_kernel(const float* __restrict__ seq,
                                 const float* __restrict__ Wsw, const float* __restrict__ Wsb,
                                 const float* __restrict__ psw, const float* __restrict__ psb,
                                 float* __restrict__ out) {
    __shared__ float srow[D];
    int i = blockIdx.x, e = threadIdx.x;
    if (e < D) srow[e] = seq[i * D + e];
    __syncthreads();
    float acc = Wsb[e] + psb[e];
    const float* wr = Wsw + e * D;
#pragma unroll
    for (int m = 0; m < D; ++m) acc += srow[m] * wr[m];
    const float* pw = psw + e * NBS;
#pragma unroll
    for (int k = 0; k < NBS; ++k)
        if ((i >> k) & 1) acc += pw[k];
    out[i * SEQ_E + e] = acc;
}

// ---------------- main: pair track, one WG per row i ----------------
__global__ void __launch_bounds__(256)
pair_kernel(const float* __restrict__ seqG,
            const float* __restrict__ Wp,      // [128][32][32] = W_pair_w
            const float* __restrict__ SPI,
            const float* __restrict__ SPJc,
            const float* __restrict__ PWT,
            float* __restrict__ outPair) {
    __shared__ _Float16 seqH[L * SEQH_STRIDE];   // f16 seq, all rows
    __shared__ _Float16 Tlds[P * T_STRIDE];      // T_i = seq_i x Wp, f16
    __shared__ float    srow[D];                 // seq row i, f32

    const int t = threadIdx.x;
    const int i = blockIdx.x;

    // ---- phase 0: stage seq -> f16 LDS (padded stride), and seq row i f32 ----
    const v4f* seq4 = reinterpret_cast<const v4f*>(seqG);
    for (int q = t; q < (L * D) / 4; q += 256) {       // 12 iters of float4
        int row = q >> 3, c4 = q & 7;
        v4f f = seq4[q];
        v4h h;
        h[0] = (_Float16)f[0]; h[1] = (_Float16)f[1];
        h[2] = (_Float16)f[2]; h[3] = (_Float16)f[3];
        *reinterpret_cast<v4h*>(&seqH[row * SEQH_STRIDE + c4 * 4]) = h;
    }
    if (t < D) srow[t] = seqG[i * D + t];
    __syncthreads();

    // ---- phase 1: T_i[p][n] = sum_m srow[m] * Wp[p][m][n], f16 into LDS ----
#pragma unroll
    for (int k = 0; k < 4; ++k) {
        int pn0 = 4 * t + 1024 * k;          // covers 0..4095
        int p  = pn0 >> 5;
        int n0 = pn0 & 31;
        const float* wrow = Wp + p * (D * D) + n0;
        v4f acc = {0.f, 0.f, 0.f, 0.f};
#pragma unroll
        for (int m = 0; m < D; ++m) {
            v4f w = *reinterpret_cast<const v4f*>(wrow + m * D);
            float s = srow[m];
            acc[0] += s * w[0]; acc[1] += s * w[1];
            acc[2] += s * w[2]; acc[3] += s * w[3];
        }
        v4h h;
        h[0] = (_Float16)acc[0]; h[1] = (_Float16)acc[1];
        h[2] = (_Float16)acc[2]; h[3] = (_Float16)acc[3];
        *reinterpret_cast<v4h*>(&Tlds[p * T_STRIDE + n0]) = h;
    }
    __syncthreads();

    // ---- phase 2: WMMA. wave w owns M-tile p0 = 16w; loops 24 N-tiles ----
    const int lane = t & 31;
    const int wv   = t >> 5;
    const int g    = lane >> 4;     // lane group
    const int r16  = lane & 15;
    const int p0   = wv * 16;

    // A fragment (16x32 f16): lane group g holds K in {8g..8g+7} U {16+8g..16+8g+7}
    const _Float16* arow = &Tlds[(p0 + r16) * T_STRIDE];
    v8h alo = *reinterpret_cast<const v8h*>(arow + 8 * g);
    v8h ahi = *reinterpret_cast<const v8h*>(arow + 16 + 8 * g);
    v16h a;
#pragma unroll
    for (int h = 0; h < 8; ++h) { a[h] = alo[h]; a[8 + h] = ahi[h]; }

    // per-lane SPJ' (depends only on i and this lane's 8 p-values)
    const float* cjp = SPJc + i * P + p0 + 8 * g;
    v4f cj0 = *reinterpret_cast<const v4f*>(cjp);
    v4f cj1 = *reinterpret_cast<const v4f*>(cjp + 4);

    for (int nt = 0; nt < 24; ++nt) {
        const int j = nt * 16 + r16;

        // B fragment (32x16 f16): lane = column j, halves = K {8g..8g+7} U {16+8g..16+8g+7}
        const _Float16* sb = &seqH[j * SEQH_STRIDE + 8 * g];
        v8h b0 = *reinterpret_cast<const v8h*>(sb);
        v8h b1 = *reinterpret_cast<const v8h*>(sb + 16);
        v16h b;
#pragma unroll
        for (int h = 0; h < 8; ++h) { b[h] = b0[h]; b[8 + h] = b1[h]; }

        v8f c = {};
        c = __builtin_amdgcn_wmma_f32_16x16x32_f16(false, a, false, b,
                                                   (short)0, c, false, false);

        // fused epilogue
        int rel = j - i;
        rel = rel < -32 ? -32 : (rel > 32 ? 32 : rel);
        rel += 32;
        const float* spip = SPI + j * P + p0 + 8 * g;
        v4f s0 = *reinterpret_cast<const v4f*>(spip);
        v4f s1 = *reinterpret_cast<const v4f*>(spip + 4);
        const float* pwp = PWT + rel * P + p0 + 8 * g;
        v4f w0 = *reinterpret_cast<const v4f*>(pwp);
        v4f w1 = *reinterpret_cast<const v4f*>(pwp + 4);

        v4f o0, o1;
#pragma unroll
        for (int r = 0; r < 4; ++r) {
            o0[r] = c[r]     + s0[r] + w0[r] + cj0[r];
            o1[r] = c[4 + r] + s1[r] + w1[r] + cj1[r];
        }
        float* op = outPair + ((size_t)i * L + j) * P + p0 + 8 * g;
        *reinterpret_cast<v4f*>(op)     = o0;
        *reinterpret_cast<v4f*>(op + 4) = o1;
    }
}

extern "C" void kernel_launch(void* const* d_in, const int* in_sizes, int n_in,
                              void* d_out, int out_size, void* d_ws, size_t ws_size,
                              hipStream_t stream) {
    const float* seq        = (const float*)d_in[0];
    const float* W_seq_w    = (const float*)d_in[1];
    const float* W_seq_b    = (const float*)d_in[2];
    const float* W_pair_w   = (const float*)d_in[3];
    const float* W_pair_b   = (const float*)d_in[4];
    const float* pos_seq_w  = (const float*)d_in[5];
    const float* pos_seq_b  = (const float*)d_in[6];
    const float* pos_pair_w = (const float*)d_in[7];
    const float* pos_pair_b = (const float*)d_in[8];
    const float* spi_w      = (const float*)d_in[9];
    const float* spi_b      = (const float*)d_in[10];
    const float* spj_w      = (const float*)d_in[11];
    const float* spj_b      = (const float*)d_in[12];

    float* out_seq  = (float*)d_out;                    // [384][256]
    float* out_pair = out_seq + L * SEQ_E;              // [384][384][128]

    float* SPI  = (float*)d_ws;                         // [384][128]
    float* SPJc = SPI + L * P;                          // [384][128]
    float* PWT  = SPJc + L * P;                         // [65][128]

    pwt_kernel<<<1, 256, 0, stream>>>(pos_pair_w, PWT);
    sp_kernel<<<L, P, 0, stream>>>(seq, spi_w, spi_b, spj_w, spj_b,
                                   W_pair_b, pos_pair_b, SPI, SPJc);
    seq_embed_kernel<<<L, SEQ_E, 0, stream>>>(seq, W_seq_w, W_seq_b,
                                              pos_seq_w, pos_seq_b, out_seq);
    pair_kernel<<<L, 256, 0, stream>>>(seq, W_pair_w, SPI, SPJc, PWT, out_pair);
}
